// SparseMoE_ExpertParallelism_46523085750483
// MI455X (gfx1250) — matrix operand order repaired and checked by reading, not compiled, not run on hardware
//
#include <hip/hip_runtime.h>
#include <hip/hip_bf16.h>

// ---------------- problem constants ----------------
#define Tt 1024   // tokens
#define Dd 1024   // hidden
#define Hh 4096   // ffn hidden
#define Ee 8      // experts
#define TM 16     // tokens per FFN tile (WMMA M)
#define NH 128    // H-tile width per workgroup iteration
#define BK 32     // WMMA K step (bf16)
#define WTILE (NH * BK)   // 4096 elems per W1/W3 tile buffer

typedef __attribute__((ext_vector_type(16))) __bf16 v16bf;
typedef __attribute__((ext_vector_type(8)))  float  v8f;
typedef __attribute__((ext_vector_type(4)))  int    v4i;

union Frag { uint4 q[2]; v16bf v; };

// ---- CDNA5 async global->LDS DMA (ASYNCcnt-tracked) ----
#define GLOBAL_AS __attribute__((address_space(1)))
#define LDS_AS    __attribute__((address_space(3)))

__device__ __forceinline__ void async_copy16(const void* g, void* l) {
#if __has_builtin(__builtin_amdgcn_global_load_async_to_lds_b128)
    __builtin_amdgcn_global_load_async_to_lds_b128(
        (GLOBAL_AS v4i*)g, (LDS_AS v4i*)l, 0, 0);
#else
    asm volatile("global_load_async_to_lds_b128 %0, %1, off"
                 :: "v"((unsigned)(unsigned long long)l), "v"(g) : "memory");
#endif
}

#if __has_builtin(__builtin_amdgcn_s_wait_asynccnt)
#define WAIT_ASYNC(n) __builtin_amdgcn_s_wait_asynccnt(n)
#else
#define WAIT_ASYNC(n) asm volatile("s_wait_asynccnt %0" :: "i"(n) : "memory")
#endif

// fp32 -> bf16 round-to-nearest-even
__device__ __forceinline__ unsigned short f2bf(float f) {
    unsigned int u = __builtin_bit_cast(unsigned int, f);
    u += 0x7FFFu + ((u >> 16) & 1u);
    return (unsigned short)(u >> 16);
}

// A-fragment (16x32 bf16) from row-major LDS tile (CDNA5 WMMA A layout).
__device__ __forceinline__ v16bf loadA(const unsigned short* base, int stride,
                                       int k0, int lane) {
    int row = lane & 15;
    int kb  = (lane >> 4) << 3;
    const unsigned short* p = base + row * stride + k0 + kb;
    Frag f;
    f.q[0] = *(const uint4*)(p);
    f.q[1] = *(const uint4*)(p + 16);
    return f.v;
}

// B-fragment (32x16 bf16) from column-major LDS tile ([cols][32]).
__device__ __forceinline__ v16bf loadB(const unsigned short* colmaj, int col0,
                                       int lane) {
    int col = col0 + (lane & 15);
    int kb  = (lane >> 4) << 3;
    const unsigned short* p = colmaj + col * BK + kb;
    Frag f;
    f.q[0] = *(const uint4*)(p);
    f.q[1] = *(const uint4*)(p + 16);
    return f.v;
}

// ---------------- gate: logits, top-2, softmax, expert lists ----------------
__global__ __launch_bounds__(256) void moe_gate_kernel(
    const float* __restrict__ X, const float* __restrict__ Wg,
    const float* __restrict__ bg, int* __restrict__ counts,
    int* __restrict__ toklist, float* __restrict__ twlist) {
    int lane = threadIdx.x & 31;
    int t = (blockIdx.x * blockDim.x + threadIdx.x) >> 5;  // one wave per token
    if (t >= Tt) return;

    float acc[Ee];
#pragma unroll
    for (int e = 0; e < Ee; ++e) acc[e] = 0.f;

    const float* xrow = X + (size_t)t * Dd;
    for (int i = 0; i < Dd / 32; ++i) {
        int d = lane + (i << 5);
        float x = xrow[d];
        const float* wr = Wg + (size_t)d * Ee;
#pragma unroll
        for (int e = 0; e < Ee; ++e) acc[e] += x * wr[e];
    }
#pragma unroll
    for (int off = 16; off > 0; off >>= 1) {
#pragma unroll
        for (int e = 0; e < Ee; ++e) acc[e] += __shfl_xor(acc[e], off, 32);
    }
    if (lane == 0) {
        float v[Ee];
#pragma unroll
        for (int e = 0; e < Ee; ++e) v[e] = acc[e] + bg[e];
        int i0 = 0; float v0 = v[0];
#pragma unroll
        for (int e = 1; e < Ee; ++e) if (v[e] > v0) { v0 = v[e]; i0 = e; }
        int i1 = (i0 == 0) ? 1 : 0; float v1 = v[i1];
#pragma unroll
        for (int e = 0; e < Ee; ++e)
            if (e != i0 && v[e] > v1) { v1 = v[e]; i1 = e; }
        float e1 = __expf(v1 - v0);
        float inv = 1.0f / (1.0f + e1);
        int c0 = atomicAdd(&counts[i0], 1);
        toklist[i0 * Tt + c0] = t;  twlist[i0 * Tt + c0] = inv;
        int c1 = atomicAdd(&counts[i1], 1);
        toklist[i1 * Tt + c1] = t;  twlist[i1 * Tt + c1] = e1 * inv;
    }
}

// -------- tiled transpose + fp32->bf16: src [R][C] -> dst [C][R] per z-slice --------
__global__ __launch_bounds__(256) void transpose_convert_kernel(
    const float* __restrict__ src, unsigned short* __restrict__ dst,
    int R, int C) {
    __shared__ unsigned short t[64 * 65];
    size_t mo = (size_t)blockIdx.z * R * C;
    src += mo; dst += mo;
    const int c0 = blockIdx.x << 6, r0 = blockIdx.y << 6, tid = threadIdx.x;
#pragma unroll
    for (int i = 0; i < 16; ++i) {        // 64x64 coalesced load + convert
        int idx = tid + (i << 8);
        int r = idx >> 6, c = idx & 63;
        t[r * 65 + c] = f2bf(src[(size_t)(r0 + r) * C + c0 + c]);
    }
    __syncthreads();
#pragma unroll
    for (int i = 0; i < 8; ++i) {         // transposed, packed b32 stores
        int idx = tid + (i << 8);
        int cc = idx >> 5, rr = idx & 31;
        unsigned int lo = t[(2 * rr) * 65 + cc];
        unsigned int hi = t[(2 * rr + 1) * 65 + cc];
        unsigned int* drow = (unsigned int*)(dst + (size_t)(c0 + cc) * R + r0);
        drow[rr] = lo | (hi << 16);
    }
}

// ---------------- per-expert fused SwiGLU FFN (WMMA bf16) ----------------
// CACHED: weights pre-transposed bf16 in ws; tiles DMA'd via async-to-LDS,
// W1/W3 double-buffered so next tile's DMA overlaps this tile's WMMAs.
template <bool CACHED>
__global__ __launch_bounds__(256) void moe_ffn_kernel(
    const float* __restrict__ X, const float* __restrict__ W1,
    const float* __restrict__ W2, const float* __restrict__ W3,
    const unsigned short* __restrict__ w1t,   // [E][H][D] bf16
    const unsigned short* __restrict__ w3t,   // [E][H][D] bf16
    const unsigned short* __restrict__ w2t,   // [E][D][H] bf16
    const int* __restrict__ counts, const int* __restrict__ toklist,
    const float* __restrict__ twlist, float* __restrict__ out) {
    extern __shared__ char smem[];
    unsigned short* sXs = (unsigned short*)smem;       // [16][1024]   32 KB
    unsigned short* sW1 = sXs + TM * Dd;               // 2x[128][32]  16 KB (col-major)
    unsigned short* sW3 = sW1 + 2 * WTILE;             // 2x[128][32]  16 KB
    unsigned short* sH  = sW3 + 2 * WTILE;             // [16][128]     4 KB
    unsigned short* sW2 = sH + TM * NH;                // [1024][32]   64 KB (col-major)

    __shared__ int   stok[TM];
    __shared__ float sw[TM];

    const int e    = blockIdx.y;
    const int tile = blockIdx.x;
    const int cnt  = counts[e];
    if (tile * TM >= cnt) return;  // uniform exit, EXEC stays full

    const int tid  = threadIdx.x;
    const int lane = tid & 31;
    const int wv   = tid >> 5;     // 8 waves

    if (tid < TM) {
        int idx = tile * TM + tid;
        bool valid = idx < cnt;
        stok[tid] = valid ? toklist[e * Tt + idx] : 0;
        sw[tid]   = valid ? twlist[e * Tt + idx] : 0.0f;  // pad rows -> exact 0
    }
    __syncthreads();

    // stage scaled token tile: sXs[r][c] = bf16(w[r] * X[tok[r]][c])
#pragma unroll 4
    for (int i = 0; i < (TM * Dd) / 256; ++i) {
        int idx = tid + (i << 8);
        int r = idx >> 10, c = idx & (Dd - 1);
        sXs[idx] = f2bf(X[(size_t)stok[r] * Dd + c] * sw[r]);
    }

    const float* W1e = W1 + (size_t)e * Dd * Hh;
    const float* W3e = W3 + (size_t)e * Dd * Hh;
    const float* W2e = W2 + (size_t)e * Hh * Dd;
    const unsigned short* g1 = w1t + (size_t)e * Hh * Dd;
    const unsigned short* g3 = w3t + (size_t)e * Hh * Dd;
    const unsigned short* g2 = w2t + (size_t)e * Dd * Hh;

    // persistent output accumulators: 8 x (16x16) f32 tiles per wave = 16x1024
    v8f accy[8];
#pragma unroll
    for (int n = 0; n < 8; ++n) accy[n] = (v8f)(0.0f);

    for (int ht = 0; ht < Hh / NH; ++ht) {     // 32 H-tiles
        const int hbase = ht * NH;
        v8f acc1 = (v8f)(0.0f), acc3 = (v8f)(0.0f);

        // 4 async-DMA instructions per wave per (W1+W3) tile pair
        auto issueW13 = [&](int kt) {
            const int k0 = kt * BK;
            unsigned short* d1 = sW1 + (kt & 1) * WTILE;
            unsigned short* d3 = sW3 + (kt & 1) * WTILE;
#pragma unroll
            for (int j = 0; j < 2; ++j) {
                int idx = tid + (j << 8);          // 0..511
                int r = idx >> 2, sub = (idx & 3) << 3;
                async_copy16(g1 + (size_t)(hbase + r) * Dd + k0 + sub,
                             d1 + r * BK + sub);
            }
#pragma unroll
            for (int j = 0; j < 2; ++j) {
                int idx = tid + (j << 8);
                int r = idx >> 2, sub = (idx & 3) << 3;
                async_copy16(g3 + (size_t)(hbase + r) * Dd + k0 + sub,
                             d3 + r * BK + sub);
            }
        };

        if constexpr (CACHED) {
            __syncthreads();      // prior readers of buffer 0 finished
            issueW13(0);          // prologue fill
        }

        // ---- up/gate projections over K = D ----
        for (int kt = 0; kt < Dd / BK; ++kt) {   // 32 steps
            const unsigned short* c1;
            const unsigned short* c3;
            if constexpr (CACHED) {
                if (kt + 1 < Dd / BK) {
                    __syncthreads();     // readers of buf[(kt+1)&1] done
                    issueW13(kt + 1);    // prefetch next tile (overlaps WMMAs)
                    WAIT_ASYNC(4);       // current tile landed; next in flight
                } else {
                    WAIT_ASYNC(0);
                }
                __syncthreads();         // all waves' current tile visible
                c1 = sW1 + (kt & 1) * WTILE;
                c3 = sW3 + (kt & 1) * WTILE;
            } else {
                const int k0 = kt * BK;
                __syncthreads();
#pragma unroll 4
                for (int i = 0; i < 16; ++i) {   // convert-in-kernel fallback
                    int idx = tid + (i << 8);
                    int r = idx >> 7, c = idx & (NH - 1);
                    sW1[c * BK + r] = f2bf(W1e[(size_t)(k0 + r) * Hh + hbase + c]);
                    sW3[c * BK + r] = f2bf(W3e[(size_t)(k0 + r) * Hh + hbase + c]);
                }
                __syncthreads();
                c1 = sW1;
                c3 = sW3;
            }
            v16bf a  = loadA(sXs, Dd, kt * BK, lane);
            v16bf b1 = loadB(c1, wv << 4, lane);
            v16bf b3 = loadB(c3, wv << 4, lane);
            acc1 = __builtin_amdgcn_wmma_f32_16x16x32_bf16(
                false, a, false, b1, (short)0, acc1, false, false);
            acc3 = __builtin_amdgcn_wmma_f32_16x16x32_bf16(
                false, a, false, b3, (short)0, acc3, false, false);
        }

        // ---- h = silu(acc1) * acc3 -> bf16 in sH[16][128] ----
        __syncthreads();  // previous ht's sH readers done
#pragma unroll
        for (int r = 0; r < 8; ++r) {
            float g  = acc1[r];
            float hv = (g / (1.0f + __expf(-g))) * acc3[r];
            int M = ((lane >> 4) << 3) + r;
            int N = (wv << 4) + (lane & 15);
            sH[M * NH + N] = f2bf(hv);
        }

        // ---- down projection: accy += h(16x128) @ W2[hbase:+128, 0:1024] ----
        for (int kk = 0; kk < NH / BK; ++kk) {   // 4 steps
            const int h0 = hbase + kk * BK;
            __syncthreads();
            if constexpr (CACHED) {
#pragma unroll
                for (int j = 0; j < 16; ++j) {     // 1024x32 bf16 tile of W2t
                    int idx = tid + (j << 8);      // 0..4095
                    int r = idx >> 2, sub = (idx & 3) << 3;
                    async_copy16(g2 + (size_t)r * Hh + h0 + sub,
                                 sW2 + r * BK + sub);
                }
                WAIT_ASYNC(0);
            } else {
#pragma unroll 8
                for (int i = 0; i < 128; ++i) {
                    int idx = tid + (i << 8);
                    int r = idx >> 10, c = idx & (Dd - 1);
                    sW2[c * BK + r] = f2bf(W2e[(size_t)(h0 + r) * Dd + c]);
                }
            }
            __syncthreads();
            v16bf ah = loadA(sH, NH, kk * BK, lane);
#pragma unroll
            for (int n = 0; n < 8; ++n) {
                v16bf b2 = loadB(sW2, (wv << 7) + (n << 4), lane);
                accy[n] = __builtin_amdgcn_wmma_f32_16x16x32_bf16(
                    false, ah, false, b2, (short)0, accy[n], false, false);
            }
        }
    }

    // ---- scatter-add into out (a token may also live in another expert) ----
#pragma unroll
    for (int n = 0; n < 8; ++n) {
        int N = (wv << 7) + (n << 4) + (lane & 15);
#pragma unroll
        for (int r = 0; r < 8; ++r) {
            int M = ((lane >> 4) << 3) + r;
            atomicAdd(&out[(size_t)stok[M] * Dd + N], accy[n][r]);
        }
    }
}

// ---------------- host entry ----------------
extern "C" void kernel_launch(void* const* d_in, const int* in_sizes, int n_in,
                              void* d_out, int out_size, void* d_ws, size_t ws_size,
                              hipStream_t stream) {
    const float* X  = (const float*)d_in[0];
    const float* Wg = (const float*)d_in[1];
    const float* bg = (const float*)d_in[2];
    const float* W1 = (const float*)d_in[3];
    const float* W2 = (const float*)d_in[4];  // dict order: W1, W2, W3
    const float* W3 = (const float*)d_in[5];
    float* out = (float*)d_out;

    // ws layout: [E counts][E*T tok][E*T tw] (<=128KB) | bf16 W1t | W3t | W2t
    int*   counts  = (int*)d_ws;
    int*   toklist = counts + 16;
    float* twlist  = (float*)(toklist + Ee * Tt);

    const size_t wElems = (size_t)Ee * Dd * Hh;           // 33.5M elems each
    const size_t listBytes = 131072;                      // padded list region
    unsigned short* w1t = (unsigned short*)((char*)d_ws + listBytes);
    unsigned short* w3t = w1t + wElems;
    unsigned short* w2t = w3t + wElems;
    const bool cached = ws_size >= listBytes + 3 * wElems * sizeof(unsigned short);

    (void)hipMemsetAsync(counts, 0, 16 * sizeof(int), stream);
    (void)hipMemcpyAsync(out, X, (size_t)Tt * Dd * sizeof(float),
                         hipMemcpyDeviceToDevice, stream);  // residual

    moe_gate_kernel<<<Tt / 8, 256, 0, stream>>>(X, Wg, bg, counts, toklist, twlist);

    size_t smem = (size_t)(TM * Dd + 4 * WTILE + TM * NH + Dd * BK)
                  * sizeof(unsigned short);  // 135168 B
    dim3 grid(Tt / TM, Ee);  // worst case: one expert owns every token

    if (cached) {
        // one-time per-launch transpose+convert: [D][H]->[H][D], [H][D]->[D][H]
        transpose_convert_kernel<<<dim3(Hh / 64, Dd / 64, Ee), 256, 0, stream>>>(
            W1, w1t, Dd, Hh);
        transpose_convert_kernel<<<dim3(Hh / 64, Dd / 64, Ee), 256, 0, stream>>>(
            W3, w3t, Dd, Hh);
        transpose_convert_kernel<<<dim3(Dd / 64, Hh / 64, Ee), 256, 0, stream>>>(
            W2, w2t, Hh, Dd);
        moe_ffn_kernel<true><<<grid, 256, smem, stream>>>(
            X, W1, W2, W3, w1t, w3t, w2t, counts, toklist, twlist, out);
    } else {
        moe_ffn_kernel<false><<<grid, 256, smem, stream>>>(
            X, W1, W2, W3, w1t, w3t, w2t, counts, toklist, twlist, out);
    }
}
